// BaseTransformer_59923383714324
// MI455X (gfx1250) — matrix operand order repaired and checked by reading, not compiled, not run on hardware
//
#include <hip/hip_runtime.h>
#include <hip/hip_bf16.h>
#include <math.h>

typedef __attribute__((ext_vector_type(16))) _Float16 v16h;
typedef __attribute__((ext_vector_type(8)))  _Float16 v8h;
typedef __attribute__((ext_vector_type(8)))  float    v8f;

#define NTOK   393
#define BATCH  16
#define DIM    512
#define HEADS  8
#define DH     64
#define FCDIM  2048
#define DEPTH  4
#define MROWS  (BATCH * NTOK)   // 6288 = 393 tiles of 16

// ---------------------------------------------------------------------------
// Embedding: per-agent projections + sinusoidal PEs + spec tokens -> xf32
// ---------------------------------------------------------------------------
__global__ void embed_kernel(const float* __restrict__ x_in,   // [16,128,29]
                             const float* __restrict__ spec,   // [16,9]
                             const float* __restrict__ Wc, const float* __restrict__ bc,
                             const float* __restrict__ Wb, const float* __restrict__ bb,
                             const float* __restrict__ Wt, const float* __restrict__ bt,
                             const float* __restrict__ specW, const float* __restrict__ specB,
                             float* __restrict__ x)             // [B*393, 512]
{
    int row = blockIdx.x;              // 0 .. B*393-1
    int b = row / NTOK, t = row % NTOK;
    float* xr = x + (size_t)row * DIM;
    for (int d = threadIdx.x; d < DIM; d += blockDim.x) {
        float acc;
        if (t < 384) {
            int agent = t >> 7, tim = t & 127;
            const float* feats = x_in + ((size_t)b * 128 + tim) * 29;
            if (agent == 0) {
                acc = bc[d];
                #pragma unroll
                for (int f = 0; f < 9;  ++f) acc += feats[f]      * Wc[f * DIM + d];
            } else if (agent == 1) {
                acc = bb[d];
                #pragma unroll
                for (int f = 0; f < 12; ++f) acc += feats[9 + f]  * Wb[f * DIM + d];
            } else {
                acc = bt[d];
                #pragma unroll
                for (int f = 0; f < 8;  ++f) acc += feats[21 + f] * Wt[f * DIM + d];
            }
            int e = d & ~1;
            float w = expf(-logf(10000.0f) * (float)e / (float)DIM);
            float aa = (float)agent * w, tt = (float)tim * w;
            acc += (d & 1) ? (cosf(aa) + cosf(tt)) : (sinf(aa) + sinf(tt));
        } else {
            int f = t - 384;
            acc = spec[b * 9 + f] * specW[f * DIM + d] + specB[f * DIM + d];
        }
        xr[d] = acc;
    }
}

// ---------------------------------------------------------------------------
// LayerNorm row -> f16 output (for WMMA A operand)
// ---------------------------------------------------------------------------
__global__ void ln_f16_kernel(const float* __restrict__ x,
                              const float* __restrict__ g, const float* __restrict__ bta,
                              _Float16* __restrict__ out)
{
    __shared__ float red[256];
    int row = blockIdx.x, tid = threadIdx.x;
    const float* xr = x + (size_t)row * DIM;
    float s = 0.f;
    for (int d = tid; d < DIM; d += 256) s += xr[d];
    red[tid] = s; __syncthreads();
    for (int off = 128; off; off >>= 1) { if (tid < off) red[tid] += red[tid + off]; __syncthreads(); }
    float mu = red[0] * (1.0f / DIM);
    __syncthreads();
    float v = 0.f;
    for (int d = tid; d < DIM; d += 256) { float dd = xr[d] - mu; v += dd * dd; }
    red[tid] = v; __syncthreads();
    for (int off = 128; off; off >>= 1) { if (tid < off) red[tid] += red[tid + off]; __syncthreads(); }
    float rs = rsqrtf(red[0] * (1.0f / DIM) + 1e-5f);
    for (int d = tid; d < DIM; d += 256)
        out[(size_t)row * DIM + d] = (_Float16)((xr[d] - mu) * rs * g[d] + bta[d]);
}

// ---------------------------------------------------------------------------
// Weight convert + transpose: f32 [K,N] -> f16 [N,K]
// ---------------------------------------------------------------------------
__global__ void convt_kernel(const float* __restrict__ W, _Float16* __restrict__ Wt,
                             int K, int N)
{
    int idx = blockIdx.x * blockDim.x + threadIdx.x;
    if (idx >= K * N) return;
    int n = idx / K, k = idx % K;
    Wt[idx] = (_Float16)W[(size_t)k * N + n];
}

// ---------------------------------------------------------------------------
// WMMA GEMM: C[M,N] = A[M,K](f16) @ Wt[N,K](f16)^T  (+bias, +gelu, f16 out or
// f32 residual-accumulate). One wave computes a 16x64 strip: one A fragment
// reused across 4 N-tiles -> 4 independent WMMA accumulator chains.
// ---------------------------------------------------------------------------
__global__ void gemm_wmma_kernel(const _Float16* __restrict__ A,
                                 const _Float16* __restrict__ Wt,
                                 int M, int N, int K,
                                 const float* __restrict__ bias,
                                 _Float16* __restrict__ out16,
                                 float* __restrict__ accf32,
                                 int gelu)
{
    int wid  = threadIdx.x >> 5;
    int lane = threadIdx.x & 31;
    int strpN = blockIdx.x;                      // 64-wide N strip
    int tileM = blockIdx.y * (blockDim.x >> 5) + wid;
    int Mtiles = M >> 4;
    if (tileM >= Mtiles) return;                 // wave-uniform exit
    int hl = lane & 15;
    int hb = lane >> 4;                          // lane-half select
    const _Float16* arow = A + (size_t)(tileM * 16 + hl) * K;
    const _Float16* brow0 = Wt + (size_t)(strpN * 64 + hl) * K;
    v8f c0 = {}, c1 = {}, c2 = {}, c3 = {};
    for (int k0 = 0; k0 < K; k0 += 32) {
        int kbA = k0 + hb * 16;
        v8h alo = *(const v8h*)(arow + kbA);
        v8h ahi = *(const v8h*)(arow + kbA + 8);
        v16h av;
        #pragma unroll
        for (int t = 0; t < 8; ++t) { av[t] = alo[t]; av[t + 8] = ahi[t]; }
        int kbB = k0 + hb * 16;
        v16h bv[4];
        #pragma unroll
        for (int u = 0; u < 4; ++u) {
            const _Float16* br = brow0 + (size_t)(u * 16) * K;
            v8h blo = *(const v8h*)(br + kbB);
            v8h bhi = *(const v8h*)(br + kbB + 8);
            #pragma unroll
            for (int t = 0; t < 8; ++t) { bv[u][t] = blo[t]; bv[u][t + 8] = bhi[t]; }
        }
        c0 = __builtin_amdgcn_wmma_f32_16x16x32_f16(false, av, false, bv[0], (short)0, c0, false, false);
        c1 = __builtin_amdgcn_wmma_f32_16x16x32_f16(false, av, false, bv[1], (short)0, c1, false, false);
        c2 = __builtin_amdgcn_wmma_f32_16x16x32_f16(false, av, false, bv[2], (short)0, c2, false, false);
        c3 = __builtin_amdgcn_wmma_f32_16x16x32_f16(false, av, false, bv[3], (short)0, c3, false, false);
    }
    v8f cc[4] = { c0, c1, c2, c3 };
    #pragma unroll
    for (int u = 0; u < 4; ++u) {
        int n = strpN * 64 + u * 16 + hl;
        float bn = bias ? bias[n] : 0.0f;
        #pragma unroll
        for (int r = 0; r < 8; ++r) {
            int m = tileM * 16 + r + hb * 8;
            float v = cc[u][r] + bn;
            if (gelu) v = 0.5f * v * (1.0f + erff(v * 0.70710678118f));
            if (out16)  out16[(size_t)m * N + n] = (_Float16)v;
            if (accf32) accf32[(size_t)m * N + n] += v;
        }
    }
}

// ---------------------------------------------------------------------------
// Fused attention: dual-score WMMA + mask blend + softmax + attn-map dump
// + P@V WMMA. One wave per (16-row band, batch*head).
// ---------------------------------------------------------------------------
__global__ void attention_kernel(const _Float16* __restrict__ qkv,   // [B*393,1536]
                                 const _Float16* __restrict__ qks,   // [B*393,1024]
                                 float* __restrict__ attn_out,       // layer slab [B,H,393,393]
                                 _Float16* __restrict__ o16)         // [B*393,512]
{
    __shared__ float    sc[16][400];
    __shared__ _Float16 pr[16][416];
    const float scale = 0.125f;                  // 64^-0.5
    int lane = threadIdx.x;
    int i0 = blockIdx.x * 16;
    int bh = blockIdx.y;
    int b = bh >> 3, h = bh & 7;
    int hl = lane & 15, hb = lane >> 4;

    const _Float16* qbase  = qkv + (size_t)b * NTOK * 1536 + h * DH;
    const _Float16* kbase  = qbase + 512;
    const _Float16* vbase  = qbase + 1024;
    const _Float16* qsbase = qks + (size_t)b * NTOK * 1024 + h * DH;
    const _Float16* ksbase = qsbase + 512;

    int mi = i0 + hl;
    bool mvalid = mi < NTOK;

    // Preload A fragments for q and qs (reused across all j tiles)
    v16h aq[2], as2[2];
    #pragma unroll
    for (int ks = 0; ks < 2; ++ks) {
        int kb = ks * 32 + hb * 16;
        v8h lo = {}, hi8 = {}, lo2 = {}, hi2 = {};
        if (mvalid) {
            lo  = *(const v8h*)(qbase  + (size_t)mi * 1536 + kb);
            hi8 = *(const v8h*)(qbase  + (size_t)mi * 1536 + kb + 8);
            lo2 = *(const v8h*)(qsbase + (size_t)mi * 1024 + kb);
            hi2 = *(const v8h*)(qsbase + (size_t)mi * 1024 + kb + 8);
        }
        #pragma unroll
        for (int t = 0; t < 8; ++t) {
            aq[ks][t]  = lo[t];  aq[ks][t + 8]  = hi8[t];
            as2[ks][t] = lo2[t]; as2[ks][t + 8] = hi2[t];
        }
    }

    // Scores: 25 j-tiles of 16, each 2 k-steps x (dots, dots_s) -> two
    // independent WMMA accumulator chains per tile.
    for (int jt = 0; jt < 25; ++jt) {
        int j0 = jt * 16;
        int nj = j0 + hl;
        bool nvalid = nj < NTOK;
        v8f cd = {}, cs = {};
        #pragma unroll
        for (int ks = 0; ks < 2; ++ks) {
            int kb = ks * 32 + hb * 16;
            v8h blo = {}, bhi = {}, blo2 = {}, bhi2 = {};
            if (nvalid) {
                blo  = *(const v8h*)(kbase  + (size_t)nj * 1536 + kb);
                bhi  = *(const v8h*)(kbase  + (size_t)nj * 1536 + kb + 8);
                blo2 = *(const v8h*)(ksbase + (size_t)nj * 1024 + kb);
                bhi2 = *(const v8h*)(ksbase + (size_t)nj * 1024 + kb + 8);
            }
            v16h bk, bks;
            #pragma unroll
            for (int t = 0; t < 8; ++t) {
                bk[t]  = blo[t];  bk[t + 8]  = bhi[t];
                bks[t] = blo2[t]; bks[t + 8] = bhi2[t];
            }
            cd = __builtin_amdgcn_wmma_f32_16x16x32_f16(false, aq[ks],  false, bk,
                                                        (short)0, cd, false, false);
            cs = __builtin_amdgcn_wmma_f32_16x16x32_f16(false, as2[ks], false, bks,
                                                        (short)0, cs, false, false);
        }
        int j = j0 + hl;
        #pragma unroll
        for (int r = 0; r < 8; ++r) {
            int mrow = r + hb * 8;
            int i = i0 + mrow;
            bool same = (i < 384) && (j < 384) && ((i >> 7) == (j >> 7));
            float v = (same ? cs[r] : cd[r]) * scale;
            if (j >= NTOK) v = -1e30f;
            sc[mrow][j] = v;
        }
    }
    __syncthreads();

    // Softmax per row (lanes 0..15 own one row each), dump probs to d_out
    if (lane < 16) {
        int i = i0 + lane;
        float mx = -1e30f;
        for (int j = 0; j < 400; ++j) mx = fmaxf(mx, sc[lane][j]);
        float sum = 0.f;
        for (int j = 0; j < 400; ++j) sum += expf(sc[lane][j] - mx);
        float inv = 1.0f / sum;
        for (int j = 0; j < 416; ++j) {
            float p = (j < 400) ? expf(sc[lane][j] - mx) * inv : 0.0f;
            pr[lane][j] = (_Float16)p;
            if (i < NTOK && j < NTOK)
                attn_out[(size_t)bh * NTOK * NTOK + (size_t)i * NTOK + j] = p;
        }
    }
    __syncthreads();

    // o = P @ V : probs A-fragment hoisted per k-step, 4 independent
    // column-tile accumulators (16x64 strip per wave).
    v8f co0 = {}, co1 = {}, co2 = {}, co3 = {};
    for (int ks = 0; ks < 13; ++ks) {
        int k0 = ks * 32;
        int kbA = k0 + hb * 16;
        v16h av;
        #pragma unroll
        for (int t = 0; t < 8; ++t) {
            av[t]     = pr[hl][kbA + t];
            av[t + 8] = pr[hl][kbA + 8 + t];
        }
        v16h bv[4];
        #pragma unroll
        for (int nt = 0; nt < 4; ++nt) {
            int nn = nt * 16 + hl;
            #pragma unroll
            for (int t = 0; t < 16; ++t) {
                int row = k0 + hb * 16 + t;
                bv[nt][t] = (row < NTOK) ? vbase[(size_t)row * 1536 + nn] : (_Float16)0.0f;
            }
        }
        co0 = __builtin_amdgcn_wmma_f32_16x16x32_f16(false, av, false, bv[0], (short)0, co0, false, false);
        co1 = __builtin_amdgcn_wmma_f32_16x16x32_f16(false, av, false, bv[1], (short)0, co1, false, false);
        co2 = __builtin_amdgcn_wmma_f32_16x16x32_f16(false, av, false, bv[2], (short)0, co2, false, false);
        co3 = __builtin_amdgcn_wmma_f32_16x16x32_f16(false, av, false, bv[3], (short)0, co3, false, false);
    }
    v8f cco[4] = { co0, co1, co2, co3 };
    #pragma unroll
    for (int nt = 0; nt < 4; ++nt) {
        int nn = nt * 16 + hl;
        #pragma unroll
        for (int r = 0; r < 8; ++r) {
            int m = i0 + r + hb * 8;
            if (m < NTOK)
                o16[((size_t)b * NTOK + m) * DIM + h * DH + nn] = (_Float16)cco[nt][r];
        }
    }
}

// ---------------------------------------------------------------------------
// Head: mean-pool over tokens, LN, Linear(512 -> 20)
// ---------------------------------------------------------------------------
__global__ void head_kernel(const float* __restrict__ x,
                            const float* __restrict__ g, const float* __restrict__ bln,
                            const float* __restrict__ W, const float* __restrict__ bias,
                            float* __restrict__ out)
{
    __shared__ float pooled[DIM];
    __shared__ float red[256];
    int b = blockIdx.x, tid = threadIdx.x;
    for (int d = tid; d < DIM; d += 256) {
        float s = 0.f;
        for (int t = 0; t < NTOK; ++t) s += x[((size_t)b * NTOK + t) * DIM + d];
        pooled[d] = s * (1.0f / NTOK);
    }
    __syncthreads();
    float s = 0.f;
    for (int d = tid; d < DIM; d += 256) s += pooled[d];
    red[tid] = s; __syncthreads();
    for (int off = 128; off; off >>= 1) { if (tid < off) red[tid] += red[tid + off]; __syncthreads(); }
    float mu = red[0] * (1.0f / DIM);
    __syncthreads();
    float v = 0.f;
    for (int d = tid; d < DIM; d += 256) { float dd = pooled[d] - mu; v += dd * dd; }
    red[tid] = v; __syncthreads();
    for (int off = 128; off; off >>= 1) { if (tid < off) red[tid] += red[tid + off]; __syncthreads(); }
    float rs = rsqrtf(red[0] * (1.0f / DIM) + 1e-5f);
    __syncthreads();
    for (int d = tid; d < DIM; d += 256) pooled[d] = (pooled[d] - mu) * rs * g[d] + bln[d];
    __syncthreads();
    if (tid < 20) {
        float acc = bias[tid];
        for (int d = 0; d < DIM; ++d) acc += pooled[d] * W[d * 20 + tid];
        out[b * 20 + tid] = acc;
    }
}

// ---------------------------------------------------------------------------
// Host-side orchestration
// ---------------------------------------------------------------------------
static inline size_t align256(size_t x) { return (x + 255) & ~(size_t)255; }

extern "C" void kernel_launch(void* const* d_in, const int* in_sizes, int n_in,
                              void* d_out, int out_size, void* d_ws, size_t ws_size,
                              hipStream_t stream)
{
    (void)in_sizes; (void)n_in; (void)out_size; (void)ws_size;
    const float* x_in    = (const float*)d_in[0];
    const float* spec    = (const float*)d_in[1];
    const float* Wc      = (const float*)d_in[2];
    const float* bc      = (const float*)d_in[3];
    const float* Wb      = (const float*)d_in[4];
    const float* bb      = (const float*)d_in[5];
    const float* Wt      = (const float*)d_in[6];
    const float* bt      = (const float*)d_in[7];
    const float* specW   = (const float*)d_in[8];
    const float* specB   = (const float*)d_in[9];
    const float* ln1_g   = (const float*)d_in[10];
    const float* ln1_b   = (const float*)d_in[11];
    const float* qkv_W   = (const float*)d_in[12];
    const float* qks_W   = (const float*)d_in[13];
    const float* out_W   = (const float*)d_in[14];
    const float* out_b   = (const float*)d_in[15];
    const float* ln2_g   = (const float*)d_in[16];
    const float* ln2_b   = (const float*)d_in[17];
    const float* ff_W1   = (const float*)d_in[18];
    const float* ff_b1   = (const float*)d_in[19];
    const float* ff_W2   = (const float*)d_in[20];
    const float* ff_b2   = (const float*)d_in[21];
    const float* gen_g   = (const float*)d_in[22];
    const float* gen_bln = (const float*)d_in[23];
    const float* gen_W   = (const float*)d_in[24];
    const float* gen_b   = (const float*)d_in[25];
    float* out = (float*)d_out;

    // Workspace carve-out (~63.5 MB total)
    char* ws = (char*)d_ws;
    float*     xf32   = (float*)ws;       ws += align256((size_t)MROWS * DIM * 4);
    _Float16*  hf16   = (_Float16*)ws;    ws += align256((size_t)MROWS * DIM * 2);
    _Float16*  bigA   = (_Float16*)ws;    ws += align256((size_t)MROWS * FCDIM * 2); // qkv / ff1
    _Float16*  qksbuf = (_Float16*)ws;    ws += align256((size_t)MROWS * 1024 * 2);
    _Float16*  o16    = (_Float16*)ws;    ws += align256((size_t)MROWS * DIM * 2);
    _Float16*  wt16   = (_Float16*)ws;    ws += align256((size_t)FCDIM * DIM * 2);

    const int GEMM_BLK = 128;                 // 4 waves -> 4 M-tiles per block
    const int MT_Y = (MROWS / 16 + 3) / 4;    // 99

    embed_kernel<<<MROWS, 256, 0, stream>>>(x_in, spec, Wc, bc, Wb, bb, Wt, bt,
                                            specW, specB, xf32);

    for (int l = 0; l < DEPTH; ++l) {
        // --- attention block ---
        ln_f16_kernel<<<MROWS, 256, 0, stream>>>(xf32, ln1_g + l * DIM, ln1_b + l * DIM, hf16);

        convt_kernel<<<(DIM * 1536 + 255) / 256, 256, 0, stream>>>(
            qkv_W + (size_t)l * DIM * 1536, wt16, DIM, 1536);
        gemm_wmma_kernel<<<dim3(1536 / 64, MT_Y), GEMM_BLK, 0, stream>>>(
            hf16, wt16, MROWS, 1536, DIM, nullptr, bigA, nullptr, 0);

        convt_kernel<<<(DIM * 1024 + 255) / 256, 256, 0, stream>>>(
            qks_W + (size_t)l * DIM * 1024, wt16, DIM, 1024);
        gemm_wmma_kernel<<<dim3(1024 / 64, MT_Y), GEMM_BLK, 0, stream>>>(
            hf16, wt16, MROWS, 1024, DIM, nullptr, qksbuf, nullptr, 0);

        float* attn_slab = out + 320 + (size_t)l * BATCH * HEADS * NTOK * NTOK;
        attention_kernel<<<dim3(25, BATCH * HEADS), 32, 0, stream>>>(
            bigA, qksbuf, attn_slab, o16);

        convt_kernel<<<(DIM * DIM + 255) / 256, 256, 0, stream>>>(
            out_W + (size_t)l * DIM * DIM, wt16, DIM, DIM);
        gemm_wmma_kernel<<<dim3(DIM / 64, MT_Y), GEMM_BLK, 0, stream>>>(
            o16, wt16, MROWS, DIM, DIM, out_b + l * DIM, nullptr, xf32, 0);

        // --- feed-forward block ---
        ln_f16_kernel<<<MROWS, 256, 0, stream>>>(xf32, ln2_g + l * DIM, ln2_b + l * DIM, hf16);

        convt_kernel<<<(DIM * FCDIM + 255) / 256, 256, 0, stream>>>(
            ff_W1 + (size_t)l * DIM * FCDIM, wt16, DIM, FCDIM);
        gemm_wmma_kernel<<<dim3(FCDIM / 64, MT_Y), GEMM_BLK, 0, stream>>>(
            hf16, wt16, MROWS, FCDIM, DIM, ff_b1 + l * FCDIM, bigA, nullptr, 1);

        convt_kernel<<<(FCDIM * DIM + 255) / 256, 256, 0, stream>>>(
            ff_W2 + (size_t)l * FCDIM * DIM, wt16, FCDIM, DIM);
        gemm_wmma_kernel<<<dim3(DIM / 64, MT_Y), GEMM_BLK, 0, stream>>>(
            bigA, wt16, MROWS, DIM, FCDIM, ff_b2 + l * DIM, nullptr, xf32, 0);
    }

    head_kernel<<<BATCH, 256, 0, stream>>>(xf32, gen_g, gen_bln, gen_W, gen_b, out);
}
